// VAE_3856880631969
// MI455X (gfx1250) — matrix operand order repaired and checked by reading, not compiled, not run on hardware
//
#include <hip/hip_runtime.h>
#include <hip/hip_bf16.h>

// ---------------------------------------------------------------------------
// VAE loss on MI455X (gfx1250, wave32). Encoder GEMMs via v_wmma_f32_16x16x32_bf16.
// B=16384, D_IN=784 (K padded ->800=25*32), D_Z=100 (W0||W1 -> 200, N padded ->224=7*32),
// N_SAMP=10. Each encoder wave computes a 16x32 output slab: one A operand feeds
// two WMMA B-tiles (halves x re-reads, 2 wmma per A load). Main K-loop is fully
// branch-free; the K tail (kt=24) is hand-peeled with the high half statically zero.
// ---------------------------------------------------------------------------

typedef __bf16 bf16_t;
typedef __attribute__((ext_vector_type(16))) __bf16 v16bf;
typedef __attribute__((ext_vector_type(8)))  __bf16 v8bf;
typedef __attribute__((ext_vector_type(8)))  float  v8f;

#define BATCH   16384
#define DIN     784
#define KPAD    800      // 25 * 32
#define DZ      100
#define NPAD    224      // 7 * 32
#define NSAMP   10
#define LOG2PI  1.8378770664093453f

// ---- workspace layout (bytes) ----
#define WS_W01B   0            // 224*800*2 = 358400
#define WS_SUMS   358400       // 256 * 4  -> 359424
#define WS_ACC    359424       // 16       -> 359440
#define WS_MEAN   359440       // 400      -> 359840
#define WS_COV    359840       // 400      -> 360240
#define WS_Z      360240       // 4000     -> 364240
#define WS_Y      364240       // 31360    -> 395600 total

// ---------------------------------------------------------------------------
// K0: convert+pad W0||W1 -> bf16 [224 x 800]; zero accumulators (every call).
// ---------------------------------------------------------------------------
__global__ __launch_bounds__(256) void vae_prep_kernel(
    const float* __restrict__ W0, const float* __restrict__ W1,
    bf16_t* __restrict__ W01b, float* __restrict__ sums, float* __restrict__ acc) {
  int idx = blockIdx.x * 256 + threadIdx.x;
  if (idx < NPAD * KPAD) {
    int r = idx / KPAD, k = idx % KPAD;
    float v = 0.0f;
    if (k < DIN) {
      if (r < DZ)            v = W0[r * DIN + k];
      else if (r < 2 * DZ)   v = W1[(r - DZ) * DIN + k];
    }
    W01b[idx] = (bf16_t)v;
  }
  if (idx < 256) sums[idx] = 0.0f;
  if (idx == 0)  acc[0] = 0.0f;
}

// ---------------------------------------------------------------------------
// helpers: WMMA operand loads (documented wave32 16-bit striping)
//   A 16x32: lanes 0-15 row M=lane,   K {koff..koff+7, koff+16..koff+23}, koff=0
//            lanes 16-31 row M=lane-16, same with koff=8
//   B 32x16: lanes 0-15 col N=lane,   K same as A (koff=0);  lanes 16-31 col N=lane-16, K same as A (koff=8)
// ---------------------------------------------------------------------------
__device__ __forceinline__ v16bf load_a_full(const float* __restrict__ xrow, int k) {
  float4 f0 = *reinterpret_cast<const float4*>(xrow + k);
  float4 f1 = *reinterpret_cast<const float4*>(xrow + k + 4);
  float4 f2 = *reinterpret_cast<const float4*>(xrow + k + 16);
  float4 f3 = *reinterpret_cast<const float4*>(xrow + k + 20);
  v16bf a;
  a[0]  = (bf16_t)f0.x; a[1]  = (bf16_t)f0.y; a[2]  = (bf16_t)f0.z; a[3]  = (bf16_t)f0.w;
  a[4]  = (bf16_t)f1.x; a[5]  = (bf16_t)f1.y; a[6]  = (bf16_t)f1.z; a[7]  = (bf16_t)f1.w;
  a[8]  = (bf16_t)f2.x; a[9]  = (bf16_t)f2.y; a[10] = (bf16_t)f2.z; a[11] = (bf16_t)f2.w;
  a[12] = (bf16_t)f3.x; a[13] = (bf16_t)f3.y; a[14] = (bf16_t)f3.z; a[15] = (bf16_t)f3.w;
  return a;
}

__device__ __forceinline__ v16bf load_a_tail(const float* __restrict__ xrow, int k) {
  // low 8 K-values valid, high 8 (k+16 >= 784) statically zero-padded
  float4 f0 = *reinterpret_cast<const float4*>(xrow + k);
  float4 f1 = *reinterpret_cast<const float4*>(xrow + k + 4);
  v16bf a;
  a[0] = (bf16_t)f0.x; a[1] = (bf16_t)f0.y; a[2] = (bf16_t)f0.z; a[3] = (bf16_t)f0.w;
  a[4] = (bf16_t)f1.x; a[5] = (bf16_t)f1.y; a[6] = (bf16_t)f1.z; a[7] = (bf16_t)f1.w;
#pragma unroll
  for (int i = 8; i < 16; ++i) a[i] = (bf16_t)0.0f;
  return a;
}

__device__ __forceinline__ v16bf load_b16(const bf16_t* __restrict__ wrow, int k) {
  v8bf lo = *reinterpret_cast<const v8bf*>(wrow + k);
  v8bf hi = *reinterpret_cast<const v8bf*>(wrow + k + 16);
  v16bf b;
#pragma unroll
  for (int i = 0; i < 8; ++i) { b[i] = lo[i]; b[8 + i] = hi[i]; }
  return b;
}

// ---------------------------------------------------------------------------
// K1: encoder GEMM: one wave -> 16x32 output slab, softplus, batch-column sums.
// grid = (1024 M-tiles, 7 N-slabs), block = 32 (one wave).
// ---------------------------------------------------------------------------
__global__ __launch_bounds__(32) void vae_encoder_kernel(
    const float* __restrict__ x, const bf16_t* __restrict__ W01b,
    const float* __restrict__ b0, const float* __restrict__ b1,
    float* __restrict__ sums) {
  const int mt   = blockIdx.x;
  const int n0   = blockIdx.y * 32;    // first of two 16-wide N tiles
  const int lane = threadIdx.x;
  const int lid  = lane & 15;
  const int hi16 = lane >> 4;
  const int koff = hi16 * 8;           // A: K chunk select per half-wave
  const int kboff = hi16 * 8;          // B: K chunk select per half-wave

  const float*  xrow  = x    + (size_t)(mt * 16 + lid) * DIN;
  const bf16_t* wrow0 = W01b + (size_t)(n0 + lid)      * KPAD;
  const bf16_t* wrow1 = W01b + (size_t)(n0 + 16 + lid) * KPAD;

  v8f c0 = {0.f, 0.f, 0.f, 0.f, 0.f, 0.f, 0.f, 0.f};
  v8f c1 = {0.f, 0.f, 0.f, 0.f, 0.f, 0.f, 0.f, 0.f};

  // branch-free main loop: kt 0..23 (max chunk start 736+8+16=760 < 784)
  for (int kt = 0; kt < 24; ++kt) {
    const int k0 = kt * 32;
    v16bf a  = load_a_full(xrow, k0 + koff);
    v16bf bA = load_b16(wrow0, k0 + kboff);
    v16bf bB = load_b16(wrow1, k0 + kboff);
    c0 = __builtin_amdgcn_wmma_f32_16x16x32_bf16(false, a, false, bA, (short)0, c0, false, false);
    c1 = __builtin_amdgcn_wmma_f32_16x16x32_bf16(false, a, false, bB, (short)0, c1, false, false);
  }
  // peeled tail kt=24: A low chunk (k=768/776) valid, high chunk (784/792) zero;
  // B reads padded W01b normally.
  {
    const int k0 = 768;
    v16bf a  = load_a_tail(xrow, k0 + koff);
    v16bf bA = load_b16(wrow0, k0 + kboff);
    v16bf bB = load_b16(wrow1, k0 + kboff);
    c0 = __builtin_amdgcn_wmma_f32_16x16x32_bf16(false, a, false, bA, (short)0, c0, false, false);
    c1 = __builtin_amdgcn_wmma_f32_16x16x32_bf16(false, a, false, bB, (short)0, c1, false, false);
  }

  // ---- epilogue: bias + softplus + reduce over the 16 M-rows of each tile ----
  const int nA = n0 + lid;
  const int nB = n0 + 16 + lid;
  float biasA = 0.0f, biasB = 0.0f;
  if (nA < DZ)           biasA = b0[nA];
  else if (nA < 2 * DZ)  biasA = b1[nA - DZ];
  if (nB < DZ)           biasB = b0[nB];
  else if (nB < 2 * DZ)  biasB = b1[nB - DZ];

  float s0 = 0.0f, s1 = 0.0f;
#pragma unroll
  for (int i = 0; i < 8; ++i) {
    float vA = c0[i] + biasA;
    float vB = c1[i] + biasB;
    s0 += fmaxf(vA, 0.0f) + __logf(1.0f + __expf(-fabsf(vA)));  // stable softplus
    s1 += fmaxf(vB, 0.0f) + __logf(1.0f + __expf(-fabsf(vB)));
  }
  __shared__ float tmp0[32], tmp1[32];
  tmp0[lane] = s0; tmp1[lane] = s1;
  __syncthreads();
  if (lane < 16) {
    float t0 = s0 + tmp0[lane + 16];   // combine M rows 0-7 with 8-15
    float t1 = s1 + tmp1[lane + 16];
    if (nA < 2 * DZ) atomicAdd(&sums[nA], t0);
    if (nB < 2 * DZ) atomicAdd(&sums[nB], t1);
  }
}

// ---------------------------------------------------------------------------
// K2: mean, cov, z = mean + cov*eps
// ---------------------------------------------------------------------------
__global__ __launch_bounds__(128) void vae_mcz_kernel(
    const float* __restrict__ sums, const float* __restrict__ eps,
    float* __restrict__ mean, float* __restrict__ cov, float* __restrict__ z) {
  int j = threadIdx.x;
  if (j < DZ) {
    float m = sums[j]      * (1.0f / (float)BATCH);
    float c = sums[DZ + j] * (1.0f / (float)BATCH);
    mean[j] = m; cov[j] = c;
#pragma unroll
    for (int s = 0; s < NSAMP; ++s) z[s * DZ + j] = m + c * eps[s * DZ + j];
  }
}

// ---------------------------------------------------------------------------
// K3: y = z @ W2^T + b2   [10 x 784], trivial size -> plain dot products.
// ---------------------------------------------------------------------------
__global__ __launch_bounds__(256) void vae_y_kernel(
    const float* __restrict__ z, const float* __restrict__ W2,
    const float* __restrict__ b2, float* __restrict__ y) {
  int idx = blockIdx.x * 256 + threadIdx.x;
  if (idx < NSAMP * DIN) {
    int s = idx / DIN, d = idx % DIN;
    const float* zr = z + s * DZ;
    const float* wr = W2 + (size_t)d * DZ;
    float dot = b2[d];
#pragma unroll 4
    for (int k = 0; k < DZ; ++k) dot += zr[k] * wr[k];
    y[idx] = dot;
  }
}

// ---------------------------------------------------------------------------
// K4: decoder: per batch row b, prod_k sigmoid((2x-1)*y) for 10 samples,
// then sum of log(prod + 0.001) -> one float atomic per block.
// ---------------------------------------------------------------------------
__global__ __launch_bounds__(256) void vae_decoder_kernel(
    const float* __restrict__ x, const float* __restrict__ y,
    float* __restrict__ acc) {
  __shared__ float ys[NSAMP * DIN];   // 31360 B
  __shared__ float red[256];
  const int b = blockIdx.x, tid = threadIdx.x;

  for (int i = tid; i < NSAMP * DIN; i += 256) ys[i] = y[i];
  __syncthreads();

  const float* xr = x + (size_t)b * DIN;
  float p[NSAMP];
#pragma unroll
  for (int s = 0; s < NSAMP; ++s) p[s] = 1.0f;

  for (int k = tid; k < DIN; k += 256) {
    float tx = 2.0f * xr[k] - 1.0f;
#pragma unroll
    for (int s = 0; s < NSAMP; ++s) {
      float t = tx * ys[s * DIN + k];
      p[s] *= 1.0f / (1.0f + __expf(-t));
    }
  }

  float lsum = 0.0f;
#pragma unroll 1
  for (int s = 0; s < NSAMP; ++s) {
    red[tid] = p[s];
    __syncthreads();
    for (int off = 128; off > 0; off >>= 1) {
      if (tid < off) red[tid] *= red[tid + off];
      __syncthreads();
    }
    if (tid == 0) lsum += __logf(red[0] + 0.001f);
    __syncthreads();
  }
  if (tid == 0) atomicAdd(acc, lsum);
}

// ---------------------------------------------------------------------------
// K5: KL terms + final scalar. One wave.
// ---------------------------------------------------------------------------
__global__ __launch_bounds__(32) void vae_final_kernel(
    const float* __restrict__ mean, const float* __restrict__ cov,
    const float* __restrict__ z, const float* __restrict__ acc,
    float* __restrict__ out) {
  __shared__ float pa[NSAMP][32], pb[NSAMP][32], pl[32];
  const int l = threadIdx.x;
  float a[NSAMP], bb[NSAMP], sl = 0.0f;
#pragma unroll
  for (int s = 0; s < NSAMP; ++s) { a[s] = 0.0f; bb[s] = 0.0f; }

  for (int j = l; j < DZ; j += 32) {
    float m = mean[j], c = cov[j];
    sl += __logf(c);
    float rc = 1.0f / c;
#pragma unroll
    for (int s = 0; s < NSAMP; ++s) {
      float zz = z[s * DZ + j];
      a[s] += zz * zz;
      float d = zz - m;
      bb[s] += d * d * rc;
    }
  }
  pl[l] = sl;
#pragma unroll
  for (int s = 0; s < NSAMP; ++s) { pa[s][l] = a[s]; pb[s][l] = bb[s]; }
  __syncthreads();

  if (l == 0) {
    float slc = 0.0f;
    for (int i = 0; i < 32; ++i) slc += pl[i];
    float dkl = 0.0f;
    for (int s = 0; s < NSAMP; ++s) {
      float A = 0.0f, Bq = 0.0f;
      for (int i = 0; i < 32; ++i) { A += pa[s][i]; Bq += pb[s][i]; }
      float log_pz = -0.5f * A  - 0.5f * (float)DZ * LOG2PI;
      float log_q  = -0.5f * Bq - 0.5f * slc - 0.5f * (float)DZ * LOG2PI;
      dkl += (log_pz - log_q);
    }
    dkl *= (1.0f / (float)NSAMP);
    float E = acc[0] * (1.0f / ((float)BATCH * (float)NSAMP));
    out[0] = -(E + dkl);
  }
}

// ---------------------------------------------------------------------------
extern "C" void kernel_launch(void* const* d_in, const int* in_sizes, int n_in,
                              void* d_out, int out_size, void* d_ws, size_t ws_size,
                              hipStream_t stream) {
  const float* x   = (const float*)d_in[0];
  const float* W0  = (const float*)d_in[1];
  const float* b0  = (const float*)d_in[2];
  const float* W1  = (const float*)d_in[3];
  const float* b1  = (const float*)d_in[4];
  const float* W2  = (const float*)d_in[5];
  const float* b2  = (const float*)d_in[6];
  const float* eps = (const float*)d_in[7];
  float* out = (float*)d_out;

  char* ws = (char*)d_ws;                       // needs ~396 KB
  bf16_t* W01b = (bf16_t*)(ws + WS_W01B);
  float*  sums = (float*)(ws + WS_SUMS);
  float*  acc  = (float*)(ws + WS_ACC);
  float*  mn   = (float*)(ws + WS_MEAN);
  float*  cv   = (float*)(ws + WS_COV);
  float*  zp   = (float*)(ws + WS_Z);
  float*  yp   = (float*)(ws + WS_Y);

  // K0: pad/convert weights, zero accumulators (deterministic per call)
  vae_prep_kernel<<<(NPAD * KPAD + 255) / 256, 256, 0, stream>>>(W0, W1, W01b, sums, acc);

  // K1: encoder WMMA GEMM + softplus + batch-sum
  dim3 egrid(BATCH / 16, NPAD / 32);
  vae_encoder_kernel<<<egrid, 32, 0, stream>>>(x, W01b, b0, b1, sums);

  // K2: mean/cov/z
  vae_mcz_kernel<<<1, 128, 0, stream>>>(sums, eps, mn, cv, zp);

  // K3: y = z @ W2^T + b2
  vae_y_kernel<<<(NSAMP * DIN + 255) / 256, 256, 0, stream>>>(zp, W2, b2, yp);

  // K4: decoder product + log accumulation
  vae_decoder_kernel<<<BATCH, 256, 0, stream>>>(x, yp, acc);

  // K5: finalize scalar
  vae_final_kernel<<<1, 32, 0, stream>>>(mn, cv, zp, acc, out);
}
